// PowerLSTMModule_13554916786411
// MI455X (gfx1250) — matrix operand order repaired and checked by reading, not compiled, not run on hardware
//
#include <hip/hip_runtime.h>

// ---------------------------------------------------------------------------
// MI455X (gfx1250) implementation.
// ~38 GFLOP total; bf16 staging keeps the inter-kernel working set (~155 MB)
// inside the 192 MB L2.  Matrix math uses v_wmma_f32_16x16x32_bf16 (wave32,
// f32 accumulate).  Recurrent phase is a persistent kernel: W_hh fragments
// live in registers across all 512 steps, h lives in LDS in WMMA A-fragment
// layout, and the per-step gate-base fetch is double-buffered through the
// Tensor Data Mover (tensor_load_to_lds + s_wait_tensorcnt) so its latency
// hides behind the WMMA + pointwise work of the previous step.
// ---------------------------------------------------------------------------

typedef __attribute__((ext_vector_type(16))) __bf16 v16bf;
typedef __attribute__((ext_vector_type(8)))  __bf16 v8bf;
typedef __attribute__((ext_vector_type(4)))  __bf16 v4bf;
typedef __attribute__((ext_vector_type(8)))  float  v8f;
typedef __attribute__((ext_vector_type(4)))  unsigned int v4u;
typedef __attribute__((ext_vector_type(8)))  int v8i;
typedef __attribute__((ext_vector_type(4)))  int v4i;

#ifndef USE_TDM
#if defined(__has_builtin)
#if __has_builtin(__builtin_amdgcn_tensor_load_to_lds)
#define USE_TDM 1
#endif
#endif
#endif
#ifndef USE_TDM
#define USE_TDM 0
#endif

#define DEV __device__ __forceinline__

DEV __bf16 f2bf(float f) { return (__bf16)f; }

DEV v16bf cat8(v8bf lo, v8bf hi) {
  return __builtin_shufflevector(lo, hi, 0,1,2,3,4,5,6,7,8,9,10,11,12,13,14,15);
}

// A fragment (16x32 bf16): rows r = lane&15 of row-major A (stride lda elems).
// lane<16 holds K = kb+0..7 and kb+16..23 ; lane>=16 holds K = kb+8..15, kb+24..31.
DEV v16bf load_a_frag(const __bf16* A, int lda, int kb, int lane) {
  const int r = lane & 15, half = lane >> 4;
  const __bf16* p = A + r * lda + kb + half * 8;
  v8bf lo = *(const v8bf*)(p);
  v8bf hi = *(const v8bf*)(p + 16);
  return cat8(lo, hi);
}

// B fragment (32x16): B[k][n] = W[n][k]; W row-major [N][ldw] bf16.
// lane n = lane&15 column n0+n; lane<16 holds K = kb+0..7, kb+16..23; lane>=16 K = kb+8..15, kb+24..31.
DEV v16bf load_b_frag(const __bf16* W, int ldw, int n0, int kb, int lane) {
  const int n = lane & 15, half = lane >> 4;
  const __bf16* p = W + (size_t)(n0 + n) * ldw + kb + half * 8;
  v8bf lo = *(const v8bf*)(p);
  v8bf hi = *(const v8bf*)(p + 16);
  return cat8(lo, hi);
}

DEV v8f wmma_bf16(v16bf a, v16bf b, v8f c) {
  return __builtin_amdgcn_wmma_f32_16x16x32_bf16(false, a, false, b, (short)0, c,
                                                 false, false);
}

DEV float sigf(float x) { return 1.0f / (1.0f + __expf(-x)); }

#if USE_TDM
// TDM: copy a 2-D tile (16 rows x 400 bf16, row stride = T*400 elems) from
// global into LDS at lds_off.  D# built per CDNA5 ISA.
// This toolchain exposes the 6-arg builtin:
//   (uint32x4 g0, int32x8 g1, int32x4 g2, int32x4 g3, int32x8, i32 cpol)
DEV void tdm_load_rows(const __bf16* gsrc, unsigned lds_off, int row_stride_elems) {
  const unsigned long long ga = (unsigned long long)(uintptr_t)gsrc;
  v4u g0;
  g0[0] = 1u;                                   // count=1, no gather
  g0[1] = lds_off;                              // lds_addr (bytes)
  g0[2] = (unsigned)ga;                         // global_addr[31:0]
  g0[3] = ((unsigned)(ga >> 32) & 0x01FFFFFFu)  // global_addr[56:32]
          | 0x80000000u;                        // type = 2 ("image")
  v8i g1;
  g1[0] = 1 << 16;                              // data_size = 2 bytes
  g1[1] = 400 << 16;                            // tensor_dim0 = 400
  g1[2] = 16 << 16;                             // tensor_dim1 = 16
  g1[3] = 400 << 16;                            // tile_dim0 = 400
  g1[4] = 16;                                   // tile_dim1 = 16, tile_dim2 = 0
  g1[5] = row_stride_elems;                     // tensor_dim0_stride[31:0]
  g1[6] = 0;                                    // stride hi / dim1_stride lo
  g1[7] = 0;
  const v4i z4 = {0, 0, 0, 0};
  const v8i z8 = {0, 0, 0, 0, 0, 0, 0, 0};
  __builtin_amdgcn_tensor_load_to_lds(g0, g1, z4, z4, z8, 0);
}
#endif

// ---------------------------------------------------------------------------
// Prep: bf16-pack weights (K padded 100->128 with zeros), fold biases,
// collapse FC head: weff = fc2_W @ fc1_W, beff = fc2_W @ fc1_b + fc2_b.
// ---------------------------------------------------------------------------
__global__ void k_prep(const float* __restrict__ W_ih0, const float* __restrict__ W_hh0,
                       const float* __restrict__ b_ih0, const float* __restrict__ b_hh0,
                       const float* __restrict__ W_ih1, const float* __restrict__ W_hh1,
                       const float* __restrict__ b_ih1, const float* __restrict__ b_hh1,
                       const float* __restrict__ fc1_W, const float* __restrict__ fc1_b,
                       const float* __restrict__ fc2_W, const float* __restrict__ fc2_b,
                       __bf16* __restrict__ Wb_ih0, __bf16* __restrict__ Wb_ih1,
                       __bf16* __restrict__ Wb_hh0, __bf16* __restrict__ Wb_hh1,
                       float* __restrict__ bias0, float* __restrict__ bias1,
                       float* __restrict__ weff, float* __restrict__ beff) {
  const int gt = blockIdx.x * blockDim.x + threadIdx.x;
  const int stride = gridDim.x * blockDim.x;
  for (int i = gt; i < 400 * 64; i += stride) Wb_ih0[i] = f2bf(W_ih0[i]);
  for (int i = gt; i < 400 * 128; i += stride) {
    const int n = i >> 7, k = i & 127;
    const bool in = k < 100;
    Wb_ih1[i] = f2bf(in ? W_ih1[n * 100 + k] : 0.0f);
    Wb_hh0[i] = f2bf(in ? W_hh0[n * 100 + k] : 0.0f);
    Wb_hh1[i] = f2bf(in ? W_hh1[n * 100 + k] : 0.0f);
  }
  for (int i = gt; i < 400; i += stride) {
    bias0[i] = b_ih0[i] + b_hh0[i];
    bias1[i] = b_ih1[i] + b_hh1[i];
  }
  for (int k = gt; k < 100; k += stride) {
    float s = 0.0f;
    for (int j = 0; j < 50; ++j) s += fc2_W[j] * fc1_W[j * 100 + k];
    weff[k] = s;
  }
  if (gt == 0) {
    float s = fc2_b[0];
    for (int j = 0; j < 50; ++j) s += fc2_W[j] * fc1_b[j];
    *beff = s;
  }
}

// ---------------------------------------------------------------------------
// LayerNorm over last dim (64) + bf16 pack.  One half-wave per row,
// one float4 per lane, shfl_xor tree reduction within the 16-lane group.
// ---------------------------------------------------------------------------
__global__ void k_ln(const float* __restrict__ x, const float* __restrict__ g,
                     const float* __restrict__ b, __bf16* __restrict__ xn) {
  const int tid = threadIdx.x;
  const int wave = tid >> 5, lane = tid & 31;
  const int half = lane >> 4, l = lane & 15;
  const size_t row = (size_t)blockIdx.x * 16 + wave * 2 + half;
  const float4 v = *(const float4*)(x + row * 64 + l * 4);
  float s = v.x + v.y + v.z + v.w;
  float q = v.x * v.x + v.y * v.y + v.z * v.z + v.w * v.w;
  #pragma unroll
  for (int m = 8; m; m >>= 1) {  // stays inside each 16-lane group
    s += __shfl_xor(s, m, 32);
    q += __shfl_xor(q, m, 32);
  }
  const float mu = s * (1.0f / 64.0f);
  const float var = q * (1.0f / 64.0f) - mu * mu;
  const float rs = rsqrtf(var + 1e-5f);
  const float4 gm = *(const float4*)(g + l * 4);
  const float4 bt = *(const float4*)(b + l * 4);
  v4bf o;
  o[0] = f2bf((v.x - mu) * rs * gm.x + bt.x);
  o[1] = f2bf((v.y - mu) * rs * gm.y + bt.y);
  o[2] = f2bf((v.z - mu) * rs * gm.z + bt.z);
  o[3] = f2bf((v.w - mu) * rs * gm.w + bt.w);
  *(v4bf*)(xn + row * 64 + l * 4) = o;
}

// ---------------------------------------------------------------------------
// Time-parallel input projection GEMM: out[M][400] (bf16) =
//   A[M][K] (bf16, row-major, stride lda) @ W[400][K]^T + bias.
// Grid: x = 25 N-tiles, y = M/128 ; block = 256 (8 waves, 1 M-tile each).
// The 16xK B tile is staged once in LDS and shared by all 8 waves.
// ---------------------------------------------------------------------------
__global__ void k_gemm(const __bf16* __restrict__ A, int lda, int K,
                       const __bf16* __restrict__ W, int ldw,
                       const float* __restrict__ bias,
                       __bf16* __restrict__ out) {
  __shared__ __bf16 Bs[16][128];
  const int tid = threadIdx.x;
  const int lane = tid & 31;
  const int wave = tid >> 5;
  const int n0 = blockIdx.x * 16;
  const size_t tm = (size_t)blockIdx.y * 8 + wave;  // M tile index
  const __bf16* Arow = A + tm * 16 * (size_t)lda;

  // Stage B tile (rows n0..n0+15, K cols) into LDS, 16B chunks.
  const int kc8 = K >> 3;  // chunks per row
  for (int c = tid; c < 16 * kc8; c += 256) {
    const int n = c / kc8, o = (c - n * kc8) * 8;
    *(v8bf*)&Bs[n][o] = *(const v8bf*)(W + (size_t)(n0 + n) * ldw + o);
  }
  __syncthreads();

  const float bv = bias[n0 + (lane & 15)];
  v8f acc;
  #pragma unroll
  for (int i = 0; i < 8; ++i) acc[i] = bv;

  for (int kb = 0; kb < K; kb += 32) {
    v16bf a = load_a_frag(Arow, lda, kb, lane);
    v16bf bf = load_b_frag(&Bs[0][0], 128, 0, kb, lane);
    acc = wmma_bf16(a, bf, acc);
  }

  const int half = lane >> 4, n = lane & 15;
  const size_t base = tm * 16 * 400 + n0 + n;
  #pragma unroll
  for (int v = 0; v < 8; ++v)
    out[base + (size_t)(v + 8 * half) * 400] = f2bf(acc[v]);
}

// ---------------------------------------------------------------------------
// Persistent recurrent LSTM layer.  16 blocks x 16 batch rows, 512 threads
// (16 wave32s).  W_hh fragments preloaded into registers for all T steps.
// h kept in LDS ([16][128] bf16, zero-padded K) as the WMMA A operand.
// Gate base (xproj) is double-buffered into LDS via the TDM one step ahead.
// ---------------------------------------------------------------------------
__global__ void __launch_bounds__(512) k_lstm(
    const __bf16* __restrict__ xproj,   // [B*T][400] bf16 (bias already folded)
    const __bf16* __restrict__ Whh,     // [400][128] bf16, zero-padded K
    __bf16* __restrict__ hout,          // [B*T][128] bf16 or null
    float* __restrict__ yout,           // [B*T] or null (fused FC head)
    const float* __restrict__ weff, const float* __restrict__ beffp, int T) {
  __shared__ __bf16 xp[2][16][400];     // TDM landing buffers (gate base)
  __shared__ float  rec[16][408];       // h @ W_hh^T from WMMA
  __shared__ __bf16 hA[16][128];
  __shared__ float  cS[16][100];
  __shared__ float  weff_s[100];
  __shared__ float  yacc[16];
  __shared__ float  beff_s;

  const int tid = threadIdx.x;
  const int lane = tid & 31, wave = tid >> 5;
  const int b0 = blockIdx.x * 16;
  const int half = lane >> 4, nn = lane & 15;

  for (int i = tid; i < 16 * 128; i += 512) (&hA[0][0])[i] = f2bf(0.0f);
  for (int i = tid; i < 16 * 100; i += 512) (&cS[0][0])[i] = 0.0f;
  if (yout) {
    for (int i = tid; i < 100; i += 512) weff_s[i] = weff[i];
    if (tid == 0) beff_s = *beffp;
  }

  // 25 N-tiles over 16 waves: wave w owns tile w, plus tile 16+w if w < 9.
  const int tile0 = wave;
  const int tile1 = 16 + wave;
  const bool has1 = (wave < 9);
  v16bf B0[4], B1[4];
  #pragma unroll
  for (int ks = 0; ks < 4; ++ks) {
    B0[ks] = load_b_frag(Whh, 128, tile0 * 16, ks * 32, lane);
    if (has1) B1[ks] = load_b_frag(Whh, 128, tile1 * 16, ks * 32, lane);
  }

#if USE_TDM
  if (wave == 0)  // prologue: fetch t=0 gate rows into buffer 0
    tdm_load_rows(xproj + ((size_t)b0 * T + 0) * 400,
                  (unsigned)(uintptr_t)&xp[0][0][0], T * 400);
#endif

  for (int t = 0; t < T; ++t) {
    const int cur = t & 1;
#if USE_TDM
    if (wave == 0) __builtin_amdgcn_s_wait_tensorcnt(0);
#else
    for (int c = tid; c < 800; c += 512) {  // 16 rows x 50 16B-chunks
      const int m = c / 50, o = (c - m * 50) * 8;
      *(v8bf*)&xp[cur][m][o] =
          *(const v8bf*)(xproj + ((size_t)(b0 + m) * T + t) * 400 + o);
    }
#endif
    __syncthreads();  // xp[cur] visible; hA from prev step visible

#if USE_TDM
    if (wave == 0 && t + 1 < T)  // prefetch next step into other buffer
      tdm_load_rows(xproj + ((size_t)b0 * T + (t + 1)) * 400,
                    (unsigned)(uintptr_t)&xp[cur ^ 1][0][0], T * 400);
#endif

    // P2: rec = h_{t-1} @ W_hh^T via WMMA (K = 128, 4 steps of 32).
    {
      v8f acc0, acc1;
      #pragma unroll
      for (int i = 0; i < 8; ++i) { acc0[i] = 0.0f; acc1[i] = 0.0f; }
      #pragma unroll
      for (int ks = 0; ks < 4; ++ks) {
        v16bf a = load_a_frag(&hA[0][0], 128, ks * 32, lane);
        acc0 = wmma_bf16(a, B0[ks], acc0);
        if (has1) acc1 = wmma_bf16(a, B1[ks], acc1);
      }
      #pragma unroll
      for (int v = 0; v < 8; ++v) {
        rec[v + 8 * half][tile0 * 16 + nn] = acc0[v];
        if (has1) rec[v + 8 * half][tile1 * 16 + nn] = acc1[v];
      }
    }
    if (yout && tid < 16) yacc[tid] = 0.0f;
    __syncthreads();

    // P3: pointwise LSTM cell (i,f,g,o), fused head accumulation (ds_add_f32).
    for (int i = tid; i < 1600; i += 512) {
      const int m = i / 100, j = i - m * 100;
      const float ig = sigf((float)xp[cur][m][j]       + rec[m][j]);
      const float fg = sigf((float)xp[cur][m][100 + j] + rec[m][100 + j]);
      const float gg = tanhf((float)xp[cur][m][200 + j] + rec[m][200 + j]);
      const float og = sigf((float)xp[cur][m][300 + j] + rec[m][300 + j]);
      const float c = fg * cS[m][j] + ig * gg;
      cS[m][j] = c;
      const float h = og * tanhf(c);
      hA[m][j] = f2bf(h);
      if (hout) hout[((size_t)(b0 + m) * T + t) * 128 + j] = f2bf(h);
      if (yout) atomicAdd(&yacc[m], weff_s[j] * h);
    }
    __syncthreads();

    if (yout && tid < 16) yout[(size_t)(b0 + tid) * T + t] = yacc[tid] + beff_s;
    // next-iteration top barrier orders the y write vs. yacc re-zeroing
  }
}

// ---------------------------------------------------------------------------
extern "C" void kernel_launch(void* const* d_in, const int* in_sizes, int n_in,
                              void* d_out, int out_size, void* d_ws, size_t ws_size,
                              hipStream_t stream) {
  const float* x     = (const float*)d_in[0];
  const float* lng   = (const float*)d_in[1];
  const float* lnb   = (const float*)d_in[2];
  const float* W_ih0 = (const float*)d_in[3];
  const float* W_hh0 = (const float*)d_in[4];
  const float* b_ih0 = (const float*)d_in[5];
  const float* b_hh0 = (const float*)d_in[6];
  const float* W_ih1 = (const float*)d_in[7];
  const float* W_hh1 = (const float*)d_in[8];
  const float* b_ih1 = (const float*)d_in[9];
  const float* b_hh1 = (const float*)d_in[10];
  const float* fc1_W = (const float*)d_in[11];
  const float* fc1_b = (const float*)d_in[12];
  const float* fc2_W = (const float*)d_in[13];
  const float* fc2_b = (const float*)d_in[14];
  float* y = (float*)d_out;

  const int T = 512;
  const size_t R = 256 * 512;  // 131072 rows

  char* w = (char*)d_ws;
  size_t off = 0;
  auto alloc = [&](size_t bytes) -> char* {
    char* p = w + off;
    off += (bytes + 255) & ~(size_t)255;
    return p;
  };
  __bf16* xn     = (__bf16*)alloc(R * 64 * 2);
  __bf16* xproj  = (__bf16*)alloc(R * 400 * 2);   // reused for both layers
  __bf16* h1     = (__bf16*)alloc(R * 128 * 2);
  __bf16* Wb_ih0 = (__bf16*)alloc(400 * 64 * 2);
  __bf16* Wb_ih1 = (__bf16*)alloc(400 * 128 * 2);
  __bf16* Wb_hh0 = (__bf16*)alloc(400 * 128 * 2);
  __bf16* Wb_hh1 = (__bf16*)alloc(400 * 128 * 2);
  float*  bias0  = (float*)alloc(400 * 4);
  float*  bias1  = (float*)alloc(400 * 4);
  float*  weff   = (float*)alloc(128 * 4);
  float*  beff   = (float*)alloc(16);

  // h1 pad columns (100..127) must be zero for the layer-1 GEMM.
  (void)hipMemsetAsync(h1, 0, R * 128 * 2, stream);

  k_prep<<<64, 256, 0, stream>>>(W_ih0, W_hh0, b_ih0, b_hh0,
                                 W_ih1, W_hh1, b_ih1, b_hh1,
                                 fc1_W, fc1_b, fc2_W, fc2_b,
                                 Wb_ih0, Wb_ih1, Wb_hh0, Wb_hh1,
                                 bias0, bias1, weff, beff);

  k_ln<<<(unsigned)(R / 16), 256, 0, stream>>>(x, lng, lnb, xn);

  // Layer 0: xproj0 = LN(x) @ W_ih0^T + (b_ih0 + b_hh0)
  k_gemm<<<dim3(25, (unsigned)(R / 128)), 256, 0, stream>>>(xn, 64, 64, Wb_ih0, 64,
                                                            bias0, xproj);
  k_lstm<<<16, 512, 0, stream>>>(xproj, Wb_hh0, h1, nullptr, nullptr, nullptr, T);

  // Layer 1: xproj1 = h1 @ W_ih1^T + (b_ih1 + b_hh1)  (buffer reuse)
  k_gemm<<<dim3(25, (unsigned)(R / 128)), 256, 0, stream>>>(h1, 128, 128, Wb_ih1, 128,
                                                            bias1, xproj);
  // Layer 1 recurrence with fused (Linear∘Linear) head -> y directly.
  k_lstm<<<16, 512, 0, stream>>>(xproj, Wb_hh1, nullptr, y, weff, beff, T);
}
